// RWKV_Block_12412455485501
// MI455X (gfx1250) — compile-verified, matrix operand order checked by reading, not run on hardware
//
#include <hip/hip_runtime.h>
#include <hip/hip_bf16.h>
#include <math.h>

// ---------------- problem constants (match reference) ----------------
#define BB 4
#define LL 512
#define EE 2048
#define HH 32
#define SS 64
#define BLROWS (BB * LL)          // 2048
#define EXTRA_MAA 32
#define EXTRA_DECAY 64
#define LN_EPS 1e-5f

// ---------------- WMMA types ----------------
typedef __attribute__((ext_vector_type(16))) __bf16 v16bf;
typedef __attribute__((ext_vector_type(8)))  __bf16 v8bf;
typedef __attribute__((ext_vector_type(8)))  float  v8f;

// K-slot permutation for bf16 WMMA fragments:
// physical K  0..7  -> slots 0..7   (lane-lo, first half)
// physical K 16..23 -> slots 8..15  (lane-lo, second half)
// physical K  8..15 -> slots 16..23 (lane-hi, first half)
// physical K 24..31 -> slots 24..31 (lane-hi, second half)
// A lane's whole 16-element fragment is one contiguous 32-byte LDS read.
__device__ __forceinline__ int kslot(int k) {
    return (k < 8) ? k : (k < 16) ? (k + 8) : (k < 24) ? (k - 8) : k;
}

// ---------------- generic bf16 WMMA GEMM ----------------
// C[M,N] = epilogue( A[M,K] @ B )  with B = W (KxN) or W^T (W stored NxK).
// Block tile 128x128, K-step 32, double-buffered LDS (one barrier/k-step).
// 256 threads = 8 wave32, wave grid 4x2; each wave computes a 32x64 patch
// = 2x4 v_wmma_f32_16x16x32_bf16 tiles.
#define BM 128
#define BN 128
#define BK 32

#define ACT_NONE    0
#define ACT_TANH    1
#define ACT_SILU    2
#define ACT_SIGMOID 3
#define ACT_RELU2   4
#define ACT_DECAY   5   // v = exp(-exp(bias + acc))

__device__ __forceinline__ v8bf pack8(float4 f0, float4 f1) {
    v8bf p;
    p[0]=(__bf16)f0.x; p[1]=(__bf16)f0.y; p[2]=(__bf16)f0.z; p[3]=(__bf16)f0.w;
    p[4]=(__bf16)f1.x; p[5]=(__bf16)f1.y; p[6]=(__bf16)f1.z; p[7]=(__bf16)f1.w;
    return p;
}

__global__ __launch_bounds__(256)
void gemm_bf16_wmma(const float* __restrict__ A, const float* __restrict__ W,
                    const float* __restrict__ bias, const float* __restrict__ mul,
                    const float* __restrict__ resid, float* __restrict__ C,
                    int M, int N, int K, int wTransposed, int act)
{
    __shared__ __bf16 As[2][BM][BK];   // 16 KB, row-major, K-slot permuted
    __shared__ __bf16 Bs[2][BN][BK];   // 16 KB, col(n)-major, K-slot permuted

    const int tid  = threadIdx.x;
    const int lane = tid & 31;
    const int wave = tid >> 5;
    const int wm   = wave >> 1;     // 0..3  (M direction, 32 rows each)
    const int wn   = wave & 1;      // 0..1  (N direction, 64 cols each)

    const int rowBlock = blockIdx.x * BM;
    const int colBlock = blockIdx.y * BN;

    v8f acc[2][4] = {};

    const int lm  = lane & 15;
    const int seg = (lane < 16) ? 0 : 16;    // contiguous fragment segment

    // staging coordinates (uniform across k-steps): 16 contiguous K floats/thread
    const int srow  = tid >> 1;              // 0..127
    const int sklo  = (tid & 1) * 16;        // physical k base 0 or 16
    const int sslot = (tid & 1) * 8;         // slots {0,16} or {8,24}

    // interior blocks (all big GEMMs) take a guard-free path
    const bool full = (rowBlock + BM <= M) && (colBlock + BN <= N);

    // ---- tile stager: global fp32 -> bf16 -> LDS buffer `buf` ----
    auto stage = [&](int k0, int buf) {
        if (full && wTransposed) {
            const float* ap = A + (size_t)(rowBlock + srow) * K + k0 + sklo;
            float4 a0 = *(const float4*)(ap + 0);
            float4 a1 = *(const float4*)(ap + 4);
            float4 a2 = *(const float4*)(ap + 8);
            float4 a3 = *(const float4*)(ap + 12);
            const float* wp = W + (size_t)(colBlock + srow) * K + k0 + sklo;
            float4 b0 = *(const float4*)(wp + 0);
            float4 b1 = *(const float4*)(wp + 4);
            float4 b2 = *(const float4*)(wp + 8);
            float4 b3 = *(const float4*)(wp + 12);
            if (k0 + BK < K) {
                __builtin_prefetch(ap + BK, 0, 1);
                __builtin_prefetch(wp + BK, 0, 1);
            }
            *(v8bf*)&As[buf][srow][sslot]      = pack8(a0, a1);
            *(v8bf*)&As[buf][srow][sslot + 16] = pack8(a2, a3);
            *(v8bf*)&Bs[buf][srow][sslot]      = pack8(b0, b1);
            *(v8bf*)&Bs[buf][srow][sslot + 16] = pack8(b2, b3);
        } else {
            float4 a0 = make_float4(0.f,0.f,0.f,0.f), a1 = a0, a2 = a0, a3 = a0;
            if (rowBlock + srow < M) {
                const float* ap = A + (size_t)(rowBlock + srow) * K + k0 + sklo;
                a0 = *(const float4*)(ap + 0);
                a1 = *(const float4*)(ap + 4);
                a2 = *(const float4*)(ap + 8);
                a3 = *(const float4*)(ap + 12);
            }
            *(v8bf*)&As[buf][srow][sslot]      = pack8(a0, a1);
            *(v8bf*)&As[buf][srow][sslot + 16] = pack8(a2, a3);
            if (wTransposed) {              // W is N x K
                float4 b0 = make_float4(0.f,0.f,0.f,0.f), b1 = b0, b2 = b0, b3 = b0;
                if (colBlock + srow < N) {
                    const float* wp = W + (size_t)(colBlock + srow) * K + k0 + sklo;
                    b0 = *(const float4*)(wp + 0);
                    b1 = *(const float4*)(wp + 4);
                    b2 = *(const float4*)(wp + 8);
                    b3 = *(const float4*)(wp + 12);
                }
                *(v8bf*)&Bs[buf][srow][sslot]      = pack8(b0, b1);
                *(v8bf*)&Bs[buf][srow][sslot + 16] = pack8(b2, b3);
            } else {                        // W is K x N (small GEMMs only)
                const int kk = tid >> 3;            // 0..31
                const int nb = (tid & 7) * 16;      // 16 n-values per thread
                const int sl = kslot(kk);
                #pragma unroll
                for (int j = 0; j < 16; ++j) {
                    const int gn = colBlock + nb + j;
                    Bs[buf][nb + j][sl] =
                        (gn < N) ? (__bf16)W[(size_t)(k0 + kk) * N + gn]
                                 : (__bf16)0.f;
                }
            }
        }
    };

    // ---- double-buffered main loop: one barrier per k-step ----
    stage(0, 0);
    __syncthreads();
    int p = 0;
    for (int k0 = 0; k0 < K; k0 += BK) {
        if (k0 + BK < K)
            stage(k0 + BK, p ^ 1);          // loads for tile i+1 overlap WMMAs of tile i

        v16bf afrag[2], bfrag[4];
        #pragma unroll
        for (int mt = 0; mt < 2; ++mt)
            afrag[mt] = *(const v16bf*)&As[p][wm * 32 + mt * 16 + lm][seg];
        #pragma unroll
        for (int nt = 0; nt < 4; ++nt)
            bfrag[nt] = *(const v16bf*)&Bs[p][wn * 64 + nt * 16 + lm][seg];

        #pragma unroll
        for (int mt = 0; mt < 2; ++mt)
            #pragma unroll
            for (int nt = 0; nt < 4; ++nt)
                acc[mt][nt] = __builtin_amdgcn_wmma_f32_16x16x32_bf16(
                    false, afrag[mt], false, bfrag[nt],
                    (short)0, acc[mt][nt], false, false);
        __syncthreads();
        p ^= 1;
    }

    // ---- epilogue: bias + activation + mul + residual ----
    const int rowoff = (lane < 16) ? 0 : 8;
    #pragma unroll
    for (int mt = 0; mt < 2; ++mt) {
        #pragma unroll
        for (int nt = 0; nt < 4; ++nt) {
            const int col = colBlock + wn * 64 + nt * 16 + lm;
            #pragma unroll
            for (int i = 0; i < 8; ++i) {
                const int row = rowBlock + wm * 32 + mt * 16 + rowoff + i;
                if (full || (row < M && col < N)) {
                    float v = acc[mt][nt][i];
                    if (bias) v += bias[col];
                    if (act == ACT_TANH)         v = tanhf(v);
                    else if (act == ACT_SILU)    v = v / (1.f + expf(-v));
                    else if (act == ACT_SIGMOID) v = 1.f / (1.f + expf(-v));
                    else if (act == ACT_RELU2)  { v = fmaxf(v, 0.f); v *= v; }
                    else if (act == ACT_DECAY)   v = expf(-expf(v));
                    const size_t o = (size_t)row * N + col;
                    if (mul)   v *= mul[o];
                    if (resid) v += resid[o];
                    C[o] = v;
                }
            }
        }
    }
}

// ---------------- LayerNorm (one row of E per block) ----------------
__global__ __launch_bounds__(256)
void ln_kernel(const float* __restrict__ x, const float* __restrict__ w,
               const float* __restrict__ b, float* __restrict__ out)
{
    const int row = blockIdx.x;
    const int tid = threadIdx.x;
    const float* xr = x + (size_t)row * EE;
    __shared__ float r1[256], r2[256];
    float s = 0.f, s2 = 0.f;
    for (int e = tid; e < EE; e += 256) { float v = xr[e]; s += v; s2 += v * v; }
    r1[tid] = s; r2[tid] = s2; __syncthreads();
    for (int off = 128; off > 0; off >>= 1) {
        if (tid < off) { r1[tid] += r1[tid + off]; r2[tid] += r2[tid + off]; }
        __syncthreads();
    }
    const float m   = r1[0] * (1.f / EE);
    const float var = r2[0] * (1.f / EE) - m * m;
    const float inv = rsqrtf(var + LN_EPS);
    for (int e = tid; e < EE; e += 256)
        out[(size_t)row * EE + e] = (xr[e] - m) * inv * w[e] + b[e];
}

// ---------------- token shift: sx[l] = prev(l) - xn[l] ----------------
__global__ __launch_bounds__(256)
void shift_kernel(const float* __restrict__ xn, const float* __restrict__ state,
                  const int* __restrict__ ip, int sel, float* __restrict__ sx)
{
    const int row = blockIdx.x;              // b*L + l
    const int b = row / LL, l = row % LL;
    const int i = ip[0];
    const int srow = (2 + SS) * i + sel;
    const float* prev = (l == 0)
        ? (state + ((size_t)b * (2 + SS) + srow) * EE)
        : (xn + (size_t)(row - 1) * EE);
    for (int e = threadIdx.x; e < EE; e += 256) {
        const size_t o = (size_t)row * EE + e;
        sx[o] = prev[e] - xn[o];
    }
}

// ---------------- out = xn + sx * coef[e] ----------------
__global__ __launch_bounds__(256)
void mix_kernel(const float* __restrict__ xn, const float* __restrict__ sx,
                const float* __restrict__ coef, float* __restrict__ out, size_t total)
{
    for (size_t idx = (size_t)blockIdx.x * blockDim.x + threadIdx.x;
         idx < total; idx += (size_t)gridDim.x * blockDim.x) {
        const int e = (int)(idx % EE);
        out[idx] = xn[idx] + sx[idx] * coef[e];
    }
}

// ---------------- x5[j] = xn + sx*(maa_stack[j] + z[j,:] @ maa_w2[j]) ----------------
__global__ __launch_bounds__(256)
void x5_kernel(const float* __restrict__ xn, const float* __restrict__ sx,
               const float* __restrict__ z, const float* __restrict__ w2,
               const float* __restrict__ stack5, float* __restrict__ x5,
               size_t F, size_t total)
{
    for (size_t idx = (size_t)blockIdx.x * blockDim.x + threadIdx.x;
         idx < total; idx += (size_t)gridDim.x * blockDim.x) {
        const size_t row = idx / EE;
        const int    e   = (int)(idx % EE);
        const float xnv = xn[idx], sxv = sx[idx];
        const float* zr = z + row * (5 * EXTRA_MAA);
        #pragma unroll
        for (int j = 0; j < 5; ++j) {
            float a = 0.f;
            #pragma unroll 8
            for (int q = 0; q < EXTRA_MAA; ++q)
                a += zr[j * EXTRA_MAA + q] * w2[(size_t)(j * EXTRA_MAA + q) * EE + e];
            x5[(size_t)j * F + idx] = xnv + sxv * (stack5[(size_t)j * EE + e] + a);
        }
    }
}

// ---------------- WKV sequential scan: one block per (b,h) chain ----------------
// 256 threads: d = tid&63 (output col), rg = tid>>6 owns 16 state rows.
__global__ __launch_bounds__(256)
void wkv_scan(const float* __restrict__ r, const float* __restrict__ k,
              const float* __restrict__ v, const float* __restrict__ w,
              const float* __restrict__ state, const int* __restrict__ ip,
              const float* __restrict__ faaaa, float* __restrict__ att)
{
    const int bh = blockIdx.x;
    const int b = bh / HH, h = bh % HH;
    const int tid = threadIdx.x;
    const int d  = tid & 63;
    const int rg = tid >> 6;                 // 0..3
    const int i  = ip[0];

    const size_t sbase = ((size_t)b * (2 + SS) + (size_t)(2 + SS) * i + 2) * EE
                       + (size_t)h * SS * SS;
    float s[16], u[16];
    #pragma unroll
    for (int j = 0; j < 16; ++j) {
        s[j] = state[sbase + (size_t)(rg * 16 + j) * SS + d];
        u[j] = faaaa[h * SS + rg * 16 + j];
    }

    __shared__ float sh_r[64], sh_k[64], sh_w[64], sh_v[64];
    __shared__ float red[4][64];

    for (int t = 0; t < LL; ++t) {
        const size_t base = ((size_t)(b * LL + t)) * EE + (size_t)h * SS;
        if (tid < 64) {
            sh_r[tid] = r[base + tid];
            sh_k[tid] = k[base + tid];
            sh_w[tid] = w[base + tid];
            sh_v[tid] = v[base + tid];
        }
        __syncthreads();
        const float vd = sh_v[d];
        float acc = 0.f;
        #pragma unroll
        for (int j = 0; j < 16; ++j) {
            const int kk = rg * 16 + j;
            const float a = sh_k[kk] * vd;
            acc += sh_r[kk] * (u[j] * a + s[j]);
            s[j] = a + sh_w[kk] * s[j];
        }
        red[rg][d] = acc;
        __syncthreads();
        if (rg == 0)
            att[base + d] = red[0][d] + red[1][d] + red[2][d] + red[3][d];
    }
}

// ---------------- per-head groupnorm, in place; multiply by silu-gated g ----------------
__global__ __launch_bounds__(256)
void gnorm_kernel(float* __restrict__ att, const float* __restrict__ g,
                  const float* __restrict__ gw, const float* __restrict__ gb)
{
    const int row = blockIdx.x;
    const int tid = threadIdx.x;
    const int h = tid >> 3, p = tid & 7;     // 8 threads/head, 8 elems each
    __shared__ float rm[32][8], rv[32][8], mh[32], ih[32];
    const size_t base = (size_t)row * EE + h * 64 + p * 8;
    float vals[8], s = 0.f, s2 = 0.f;
    #pragma unroll
    for (int j = 0; j < 8; ++j) { vals[j] = att[base + j]; s += vals[j]; s2 += vals[j] * vals[j]; }
    rm[h][p] = s; rv[h][p] = s2; __syncthreads();
    if (p == 0) {
        float a = 0.f, c = 0.f;
        #pragma unroll
        for (int q = 0; q < 8; ++q) { a += rm[h][q]; c += rv[h][q]; }
        const float m = a * (1.f / 64.f);
        mh[h] = m; ih[h] = rsqrtf(c * (1.f / 64.f) - m * m + LN_EPS);
    }
    __syncthreads();
    const float m = mh[h], iv = ih[h];
    #pragma unroll
    for (int j = 0; j < 8; ++j) {
        const int e = h * 64 + p * 8 + j;
        const float vn = (vals[j] - m) * iv * gw[e] + gb[e];
        att[base + j] = vn * g[base + j];
    }
}

// ---------------- host-side launcher ----------------
static inline void launch_gemm(const float* A, const float* W, const float* bias,
                               const float* mul, const float* resid, float* C,
                               int M, int N, int K, int T, int act, hipStream_t st)
{
    dim3 g((M + BM - 1) / BM, (N + BN - 1) / BN);
    gemm_bf16_wmma<<<g, 256, 0, st>>>(A, W, bias, mul, resid, C, M, N, K, T, act);
}

extern "C" void kernel_launch(void* const* d_in, const int* in_sizes, int n_in,
                              void* d_out, int out_size, void* d_ws, size_t ws_size,
                              hipStream_t stream)
{
    (void)in_sizes; (void)n_in; (void)out_size; (void)ws_size;
    const float* x        = (const float*)d_in[0];
    const float* state    = (const float*)d_in[1];
    const float* ln1_w    = (const float*)d_in[2];
    const float* ln1_b    = (const float*)d_in[3];
    const float* ln2_w    = (const float*)d_in[4];
    const float* ln2_b    = (const float*)d_in[5];
    const float* maa_x    = (const float*)d_in[6];
    const float* maa_w1   = (const float*)d_in[7];
    const float* maa_w2   = (const float*)d_in[8];
    const float* maa_st   = (const float*)d_in[9];
    const float* tdec     = (const float*)d_in[10];
    const float* dec_w1   = (const float*)d_in[11];
    const float* dec_w2   = (const float*)d_in[12];
    const float* faaaa    = (const float*)d_in[13];
    const float* Wr       = (const float*)d_in[14];
    const float* Wk       = (const float*)d_in[15];
    const float* Wv       = (const float*)d_in[16];
    const float* Wg       = (const float*)d_in[17];
    const float* Wo       = (const float*)d_in[18];
    const float* gn_w     = (const float*)d_in[19];
    const float* gn_b     = (const float*)d_in[20];
    const float* f_maa_k  = (const float*)d_in[21];
    const float* f_maa_r  = (const float*)d_in[22];
    const float* fWk      = (const float*)d_in[23];
    const float* fWr      = (const float*)d_in[24];
    const float* fWv      = (const float*)d_in[25];
    const int*   ip       = (const int*)d_in[26];
    float* out = (float*)d_out;

    const size_t F = (size_t)BLROWS * EE;    // 4,194,304 floats per buffer
    float* ws = (float*)d_ws;
    float* xn  = ws + 0 * F;                 // later: att (scan output)
    float* sx  = ws + 1 * F;                 // later: x after attention
    float* tb  = ws + 2 * F;                 // tmix -> decay w -> xn2
    float* x5  = ws + 3 * F;                 // x5_0..x5_4 in 3..7; later xk/xr/rr/kk
    float* rb  = ws + 8 * F;                 // r; later sx2
    float* kb  = ws + 9 * F;
    float* vb  = ws + 10 * F;
    float* gb  = ws + 11 * F;
    float* z   = ws + 12 * F;                // (BL,160)
    float* d1  = z + (size_t)BLROWS * 5 * EXTRA_MAA;   // (BL,64)

    const size_t total = F;
    const int ewGrid = 4096;

    // --- attention half ---
    ln_kernel<<<BLROWS, 256, 0, stream>>>(x, ln1_w, ln1_b, xn);
    shift_kernel<<<BLROWS, 256, 0, stream>>>(xn, state, ip, 1, sx);
    mix_kernel<<<ewGrid, 256, 0, stream>>>(xn, sx, maa_x, tb, total);
    launch_gemm(tb, maa_w1, nullptr, nullptr, nullptr, z,
                BLROWS, 5 * EXTRA_MAA, EE, 0, ACT_TANH, stream);
    x5_kernel<<<ewGrid, 256, 0, stream>>>(xn, sx, z, maa_w2, maa_st, x5, F, total);
    launch_gemm(x5 + 1 * F, dec_w1, nullptr, nullptr, nullptr, d1,
                BLROWS, EXTRA_DECAY, EE, 0, ACT_TANH, stream);
    launch_gemm(d1, dec_w2, tdec, nullptr, nullptr, tb,
                BLROWS, EE, EXTRA_DECAY, 0, ACT_DECAY, stream);
    launch_gemm(x5 + 3 * F, Wr, nullptr, nullptr, nullptr, rb, BLROWS, EE, EE, 1, ACT_NONE, stream);
    launch_gemm(x5 + 0 * F, Wk, nullptr, nullptr, nullptr, kb, BLROWS, EE, EE, 1, ACT_NONE, stream);
    launch_gemm(x5 + 2 * F, Wv, nullptr, nullptr, nullptr, vb, BLROWS, EE, EE, 1, ACT_NONE, stream);
    launch_gemm(x5 + 4 * F, Wg, nullptr, nullptr, nullptr, gb, BLROWS, EE, EE, 1, ACT_SILU, stream);
    wkv_scan<<<BB * HH, 256, 0, stream>>>(rb, kb, vb, tb, state, ip, faaaa, xn);
    gnorm_kernel<<<BLROWS, 256, 0, stream>>>(xn, gb, gn_w, gn_b);
    launch_gemm(xn, Wo, nullptr, nullptr, x, sx, BLROWS, EE, EE, 1, ACT_NONE, stream);

    // --- ffn half (channel-mix) ---
    ln_kernel<<<BLROWS, 256, 0, stream>>>(sx, ln2_w, ln2_b, tb);
    shift_kernel<<<BLROWS, 256, 0, stream>>>(tb, state, ip, 0, rb);
    mix_kernel<<<ewGrid, 256, 0, stream>>>(tb, rb, f_maa_k, x5 + 0 * F, total);  // xk
    mix_kernel<<<ewGrid, 256, 0, stream>>>(tb, rb, f_maa_r, x5 + 1 * F, total);  // xr
    launch_gemm(x5 + 1 * F, fWr, nullptr, nullptr, nullptr, x5 + 2 * F,
                BLROWS, EE, EE, 1, ACT_SIGMOID, stream);                          // rr
    launch_gemm(x5 + 0 * F, fWk, nullptr, nullptr, nullptr, x5 + 3 * F,
                BLROWS, EE, EE, 1, ACT_RELU2, stream);                            // kk
    launch_gemm(x5 + 3 * F, fWv, nullptr, x5 + 2 * F, sx, out,
                BLROWS, EE, EE, 1, ACT_NONE, stream);                             // out
}
